// GATImageClassifier_89232240542456
// MI455X (gfx1250) — compile-verified, matrix operand order checked by reading, not run on hardware
//
#include <hip/hip_runtime.h>
#include <hip/hip_bf16.h>

// ---------------- types for WMMA ----------------
typedef __attribute__((ext_vector_type(16))) __bf16 v16bf;
typedef __attribute__((ext_vector_type(8)))  __bf16 v8bf;
typedef __attribute__((ext_vector_type(8)))  float  v8f;

#define HC 512          // H*C
#define HEADS 8
#define CH 64
#define GRAPHS 64
#define OUTC 10
#define ENC_NEG_INF 0x007FFFFFu   // monotone-uint encoding of -inf

// ---------------- helpers ----------------
__device__ __forceinline__ unsigned fenc(float f) {
    unsigned u = __float_as_uint(f);
    return u ^ (((unsigned)((int)u >> 31)) | 0x80000000u);
}
__device__ __forceinline__ float fdec(unsigned e) {
    unsigned u = (e & 0x80000000u) ? (e ^ 0x80000000u) : ~e;
    return __uint_as_float(u);
}
__device__ __forceinline__ unsigned short f2bf(float f) {  // RNE f32->bf16
    unsigned u = __float_as_uint(f);
    unsigned r = u + 0x7fffu + ((u >> 16) & 1u);
    return (unsigned short)(r >> 16);
}
__device__ __forceinline__ float lrelu(float x) { return x > 0.f ? x : 0.2f * x; }

// ---------------- generic fill ----------------
__global__ void fill_u32(unsigned* __restrict__ p, unsigned v, long count) {
    long gid = (long)blockIdx.x * blockDim.x + threadIdx.x;
    if (gid < count) p[gid] = v;
}

// ---------------- W2 transpose+convert: W2[in][out] f32 -> w2t[out][in] bf16 ----------------
__global__ void prep_w2t(const float* __restrict__ W2, unsigned short* __restrict__ w2t) {
    int gid = blockIdx.x * blockDim.x + threadIdx.x;
    if (gid >= HC * HC) return;
    int o = gid >> 9, i = gid & (HC - 1);
    w2t[gid] = f2bf(W2[(size_t)i * HC + o]);
}

// ---------------- layer-1 projection: h = x[N,3] @ W1[3,512] ----------------
__global__ void gemm1_proj(const float* __restrict__ x, const float* __restrict__ W1,
                           float* __restrict__ h, int n) {
    long gid = (long)blockIdx.x * blockDim.x + threadIdx.x;
    if (gid >= (long)n * HC) return;
    int node = (int)(gid >> 9);
    int c = (int)(gid & (HC - 1));
    const float* xr = x + (size_t)node * 3;
    h[gid] = xr[0] * W1[c] + xr[1] * W1[HC + c] + xr[2] * W1[2 * HC + c];
}

// ---------------- per-node attention logits: als/ald[N,8] ----------------
__global__ void node_logits(const float* __restrict__ hfeat,
                            const float* __restrict__ a_s, const float* __restrict__ a_d,
                            float* __restrict__ als, float* __restrict__ ald, int n) {
    long gid = (long)blockIdx.x * blockDim.x + threadIdx.x;
    if (gid >= (long)n * HEADS) return;
    int node = (int)(gid >> 3), hd = (int)(gid & 7);
    const float* hp  = hfeat + (size_t)node * HC + hd * CH;
    const float* asp = a_s + hd * CH;
    const float* adp = a_d + hd * CH;
    float s = 0.f, d = 0.f;
    #pragma unroll 8
    for (int c = 0; c < CH; ++c) { float hv = hp[c]; s += hv * asp[c]; d += hv * adp[c]; }
    als[gid] = s;
    ald[gid] = d;
}

// ---------------- edge pass A: segment max via monotone-uint atomicMax ----------------
__global__ void edge_max(const float* __restrict__ als, const float* __restrict__ ald,
                         const int* __restrict__ ei, int E, int EP,
                         unsigned* __restrict__ mmax) {
    long gid = (long)blockIdx.x * blockDim.x + threadIdx.x;
    if (gid >= (long)EP * HEADS) return;
    int e = (int)(gid >> 3), hd = (int)(gid & 7);
    int s, d;
    if (e < E) { s = ei[e]; d = ei[E + e]; } else { s = d = e - E; }
    float sc = lrelu(als[(size_t)s * HEADS + hd] + ald[(size_t)d * HEADS + hd]);
    atomicMax(&mmax[(size_t)d * HEADS + hd], fenc(sc));
}

// ---------------- edge pass B: exp(e - max), store ex, segment-sum denominator ----------------
__global__ void edge_exp(const float* __restrict__ als, const float* __restrict__ ald,
                         const int* __restrict__ ei, int E, int EP,
                         const unsigned* __restrict__ mmax,
                         float* __restrict__ exbuf, float* __restrict__ den) {
    long gid = (long)blockIdx.x * blockDim.x + threadIdx.x;
    if (gid >= (long)EP * HEADS) return;
    int e = (int)(gid >> 3), hd = (int)(gid & 7);
    int s, d;
    if (e < E) { s = ei[e]; d = ei[E + e]; } else { s = d = e - E; }
    float sc = lrelu(als[(size_t)s * HEADS + hd] + ald[(size_t)d * HEADS + hd]);
    float m  = fdec(mmax[(size_t)d * HEADS + hd]);
    float ex = __expf(sc - m);
    exbuf[gid] = ex;
    atomicAdd(&den[(size_t)d * HEADS + hd], ex);
}

// ---------------- edge pass C: agg[dst] += alpha * h[src], float4 vectorized ----------------
__global__ void edge_agg(const float* __restrict__ exbuf, const float* __restrict__ den,
                         const float* __restrict__ hfeat, const int* __restrict__ ei,
                         int E, int EP, float* __restrict__ agg) {
    long gid = (long)blockIdx.x * blockDim.x + threadIdx.x;
    if (gid >= (long)EP * (HC / 4)) return;
    int e = (int)(gid >> 7);
    int q = (int)(gid & 127) << 2;      // feature base, multiple of 4
    int hd = q >> 6;
    int s, d;
    if (e < E) { s = ei[e]; d = ei[E + e]; } else { s = d = e - E; }
    float alpha = exbuf[(size_t)e * HEADS + hd] / den[(size_t)d * HEADS + hd];
    float4 hv = *(const float4*)(hfeat + (size_t)s * HC + q);
    float* dst = agg + (size_t)d * HC + q;
    atomicAdd(dst + 0, alpha * hv.x);
    atomicAdd(dst + 1, alpha * hv.y);
    atomicAdd(dst + 2, alpha * hv.z);
    atomicAdd(dst + 3, alpha * hv.w);
}

// ---------------- bias + ELU -> bf16 (feeds WMMA GEMM) ----------------
__global__ void bias_elu_bf16(const float* __restrict__ agg, const float* __restrict__ b,
                              unsigned short* __restrict__ xo, long total) {
    long gid = (long)blockIdx.x * blockDim.x + threadIdx.x;
    if (gid >= total) return;
    float v = agg[gid] + b[gid & (HC - 1)];
    v = v > 0.f ? v : (__expf(v) - 1.f);
    xo[gid] = f2bf(v);
}

// ---------------- bias + ELU in-place f32 ----------------
__global__ void bias_elu_f32(float* __restrict__ agg, const float* __restrict__ b, long total) {
    long gid = (long)blockIdx.x * blockDim.x + threadIdx.x;
    if (gid >= total) return;
    float v = agg[gid] + b[gid & (HC - 1)];
    agg[gid] = v > 0.f ? v : (__expf(v) - 1.f);
}

// ---------------- GEMM2: h2[N,512] f32 = x1(bf16) @ W2 via v_wmma_f32_16x16x32_bf16 ----------------
// Block: 256 threads = 8 waves. Block tile: 16 rows x 128 cols (one 16x16 tile per wave).
// A-tile (16 rows x K=512, bf16 = 16KB) staged in LDS once, shared by all waves.
// B read from L2-resident transposed weights w2t[out][in].
__global__ __launch_bounds__(256) void gemm2_wmma(const unsigned short* __restrict__ xa,
                                                  const unsigned short* __restrict__ w2t,
                                                  float* __restrict__ out, int n) {
    __shared__ unsigned short sA[16 * HC];          // 16 KB
    const int row0 = blockIdx.y * 16;
    const int tid = threadIdx.x;

    // cooperative A-tile load: 4096 dwords, 16 per thread
    {
        unsigned* sdst = (unsigned*)sA;
        #pragma unroll
        for (int i = 0; i < 16; ++i) {
            int idx = tid + i * 256;                // 0..4095 == row*256 + dword
            int row = idx >> 8;
            int gr = row0 + row; if (gr >= n) gr = n - 1;
            sdst[idx] = ((const unsigned*)(xa + (size_t)gr * HC))[idx & 255];
        }
    }
    __syncthreads();

    const int wv = tid >> 5, lane = tid & 31;
    const int n0 = blockIdx.x * 128 + wv * 16;
    const int l15 = lane & 15;
    const int khalf = (lane >> 4) << 3;             // 0 or 8 (ISA 16-bit A/B 16x32 layout)
    const unsigned short* arow = sA + l15 * HC;
    const unsigned short* brow = w2t + (size_t)(n0 + l15) * HC;

    v8f acc = {};
    #pragma unroll
    for (int k0 = 0; k0 < HC; k0 += 32) {
        const int kb = k0 + khalf;
        v8bf a0 = *(const v8bf*)(arow + kb);        // K = kb..kb+7
        v8bf a1 = *(const v8bf*)(arow + kb + 16);   // K = kb+16..kb+23
        v16bf A = __builtin_shufflevector(a0, a1, 0,1,2,3,4,5,6,7,8,9,10,11,12,13,14,15);
        v8bf b0 = *(const v8bf*)(brow + kb);
        v8bf b1 = *(const v8bf*)(brow + kb + 16);
        v16bf B = __builtin_shufflevector(b0, b1, 0,1,2,3,4,5,6,7,8,9,10,11,12,13,14,15);
        acc = __builtin_amdgcn_wmma_f32_16x16x32_bf16(false, A, false, B,
                                                      (short)0, acc, false, false);
    }

    // D layout: lanes 0-15 -> rows 0..7 (VGPR i = row i), lanes 16-31 -> rows 8..15
    const int col = n0 + l15;
    const int rbase = row0 + ((lane >> 4) << 3);
    #pragma unroll
    for (int i = 0; i < 8; ++i) {
        int r = rbase + i;
        if (r < n) out[(size_t)r * HC + col] = acc[i];
    }
}

// ---------------- pooling ----------------
__global__ void pool_sum(const float* __restrict__ x2, const int* __restrict__ batch,
                         float* __restrict__ psum, int n) {
    long gid = (long)blockIdx.x * blockDim.x + threadIdx.x;
    if (gid >= (long)n * (HC / 4)) return;
    int node = (int)(gid >> 7);
    int q = (int)(gid & 127) << 2;
    int g = batch[node];
    float4 v = *(const float4*)(x2 + (size_t)node * HC + q);
    float* dp = psum + (size_t)g * HC + q;
    atomicAdd(dp + 0, v.x); atomicAdd(dp + 1, v.y);
    atomicAdd(dp + 2, v.z); atomicAdd(dp + 3, v.w);
}

__global__ void pool_cnt(const int* __restrict__ batch, float* __restrict__ pcnt, int n) {
    long gid = (long)blockIdx.x * blockDim.x + threadIdx.x;
    if (gid >= n) return;
    atomicAdd(&pcnt[batch[gid]], 1.0f);
}

// ---------------- classifier: out[64,10] = (psum/cnt) @ Wc + bc ----------------
__global__ void classify(const float* __restrict__ psum, const float* __restrict__ pcnt,
                         const float* __restrict__ Wc, const float* __restrict__ bc,
                         float* __restrict__ out) {
    int t = threadIdx.x;
    if (t >= GRAPHS * OUTC) return;
    int g = t / OUTC, o = t % OUTC;
    float c = pcnt[g]; if (c < 1.f) c = 1.f;
    float inv = 1.f / c;
    float acc = 0.f;
    for (int k = 0; k < HC; ++k) acc += psum[(size_t)g * HC + k] * inv * Wc[(size_t)k * OUTC + o];
    out[t] = acc + bc[o];
}

// ---------------- host-side launch ----------------
static inline unsigned nblk(long total, int bs) { return (unsigned)((total + bs - 1) / bs); }

extern "C" void kernel_launch(void* const* d_in, const int* in_sizes, int n_in,
                              void* d_out, int out_size, void* d_ws, size_t ws_size,
                              hipStream_t stream) {
    const float* x    = (const float*)d_in[0];
    const int*   ei   = (const int*)  d_in[1];
    const int*   bat  = (const int*)  d_in[2];
    const float* W1   = (const float*)d_in[3];
    const float* a1s  = (const float*)d_in[4];
    const float* a1d  = (const float*)d_in[5];
    const float* b1   = (const float*)d_in[6];
    const float* W2   = (const float*)d_in[7];
    const float* a2s  = (const float*)d_in[8];
    const float* a2d  = (const float*)d_in[9];
    const float* b2   = (const float*)d_in[10];
    const float* Wc   = (const float*)d_in[11];
    const float* bc   = (const float*)d_in[12];
    float* out = (float*)d_out;

    const int N  = in_sizes[0] / 3;
    const int E  = in_sizes[1] / 2;
    const int EP = E + N;                       // edges + self loops

    // ---- carve workspace ----
    char* base = (char*)d_ws;
    size_t off = 0;
    auto carve = [&](size_t bytes) -> char* {
        char* r = base + off;
        off += (bytes + 255) & ~(size_t)255;
        return r;
    };
    float*          hbuf  = (float*)         carve((size_t)N * HC * 4);   // h1 then h2
    float*          agg   = (float*)         carve((size_t)N * HC * 4);   // aggregation / x2
    unsigned short* xabf  = (unsigned short*)carve((size_t)N * HC * 2);   // x1 bf16
    unsigned short* w2t   = (unsigned short*)carve((size_t)HC * HC * 2);  // W2^T bf16
    float*          als   = (float*)         carve((size_t)N * HEADS * 4);
    float*          ald   = (float*)         carve((size_t)N * HEADS * 4);
    unsigned*       mbuf  = (unsigned*)      carve((size_t)N * HEADS * 4);
    float*          den   = (float*)         carve((size_t)N * HEADS * 4);
    float*          exbuf = (float*)         carve((size_t)EP * HEADS * 4);
    float*          psum  = (float*)         carve((size_t)GRAPHS * HC * 4);
    float*          pcnt  = (float*)         carve((size_t)GRAPHS * 4);
    (void)ws_size; (void)n_in; (void)out_size;

    const int BS = 256;
    const long NHC = (long)N * HC;

    // ---- init ----
    fill_u32<<<nblk(NHC, BS), BS, 0, stream>>>((unsigned*)agg, 0u, NHC);
    fill_u32<<<nblk((long)N * HEADS, BS), BS, 0, stream>>>((unsigned*)den, 0u, (long)N * HEADS);
    fill_u32<<<nblk((long)N * HEADS, BS), BS, 0, stream>>>(mbuf, ENC_NEG_INF, (long)N * HEADS);
    fill_u32<<<nblk(GRAPHS * HC, BS), BS, 0, stream>>>((unsigned*)psum, 0u, GRAPHS * HC);
    fill_u32<<<1, 64, 0, stream>>>((unsigned*)pcnt, 0u, GRAPHS);
    prep_w2t<<<nblk(HC * HC, BS), BS, 0, stream>>>(W2, w2t);

    // ---- layer 1 ----
    gemm1_proj<<<nblk(NHC, BS), BS, 0, stream>>>(x, W1, hbuf, N);
    node_logits<<<nblk((long)N * HEADS, BS), BS, 0, stream>>>(hbuf, a1s, a1d, als, ald, N);
    edge_max<<<nblk((long)EP * HEADS, BS), BS, 0, stream>>>(als, ald, ei, E, EP, mbuf);
    edge_exp<<<nblk((long)EP * HEADS, BS), BS, 0, stream>>>(als, ald, ei, E, EP, mbuf, exbuf, den);
    edge_agg<<<nblk((long)EP * (HC / 4), BS), BS, 0, stream>>>(exbuf, den, hbuf, ei, E, EP, agg);
    bias_elu_bf16<<<nblk(NHC, BS), BS, 0, stream>>>(agg, b1, xabf, NHC);

    // ---- re-init for layer 2 ----
    fill_u32<<<nblk(NHC, BS), BS, 0, stream>>>((unsigned*)agg, 0u, NHC);
    fill_u32<<<nblk((long)N * HEADS, BS), BS, 0, stream>>>((unsigned*)den, 0u, (long)N * HEADS);
    fill_u32<<<nblk((long)N * HEADS, BS), BS, 0, stream>>>(mbuf, ENC_NEG_INF, (long)N * HEADS);

    // ---- layer 2: WMMA GEMM + attention ----
    {
        dim3 grid(HC / 128, (unsigned)((N + 15) / 16), 1);
        gemm2_wmma<<<grid, 256, 0, stream>>>(xabf, w2t, hbuf, N);
    }
    node_logits<<<nblk((long)N * HEADS, BS), BS, 0, stream>>>(hbuf, a2s, a2d, als, ald, N);
    edge_max<<<nblk((long)EP * HEADS, BS), BS, 0, stream>>>(als, ald, ei, E, EP, mbuf);
    edge_exp<<<nblk((long)EP * HEADS, BS), BS, 0, stream>>>(als, ald, ei, E, EP, mbuf, exbuf, den);
    edge_agg<<<nblk((long)EP * (HC / 4), BS), BS, 0, stream>>>(exbuf, den, hbuf, ei, E, EP, agg);
    bias_elu_f32<<<nblk(NHC, BS), BS, 0, stream>>>(agg, b2, NHC);

    // ---- pool + classify ----
    pool_sum<<<nblk((long)N * (HC / 4), BS), BS, 0, stream>>>(agg, bat, psum, N);
    pool_cnt<<<nblk(N, BS), BS, 0, stream>>>(bat, pcnt, N);
    classify<<<1, GRAPHS * OUTC, 0, stream>>>(psum, pcnt, Wc, bc, out);
}
